// unit_sgcn_37993280700797
// MI455X (gfx1250) — matrix operand order repaired and checked by reading, not compiled, not run on hardware
//
#include <hip/hip_runtime.h>
#include <hip/hip_bf16.h>

// ST-GCN fused unit for MI455X (gfx1250, wave32): bf16 WMMA + TDM + async-to-LDS.
// y[n,o,t,v] = sum_p ( W_p @ x[n,:,t,:] @ A_p )[o,v] + sum_p b[p,o], then BN+ReLU.
// bf16 WMMA (16x16x32) with f32 accumulation, two-stage GEMM through LDS.
// Constants (W, A^T as bf16; sum_p b as f32) are prepped once in d_ws; each WG then
// stages W with ONE tensor_load_to_lds DMA (TENSORcnt) and Ag/x with
// global_load_async_to_lds_* (ASYNCcnt).

typedef __bf16 bf16;
typedef __attribute__((ext_vector_type(16))) bf16 v16bf;
typedef __attribute__((ext_vector_type(8)))  bf16 v8bf;
typedef __attribute__((ext_vector_type(8)))  float v8f;
typedef __attribute__((ext_vector_type(4)))  unsigned int v4u;
typedef __attribute__((ext_vector_type(8)))  int v8i;
typedef __attribute__((ext_vector_type(4)))  int v4i;

union V16 { v16bf v; v8bf h[2]; };

#define NN   32
#define CIN  64
#define TT   4      // time steps per workgroup (300 = 75*4, no ragged tiles)
#define TSTEPS 300
#define VV   25
#define VP   32     // V padded to 32 (two 16-wide WMMA column tiles)
#define PP   3
#define COUT 128

// ---- d_ws layout (bytes) --------------------------------------------------
//  [0, 1024)        : BN scale[128], shift[128] (f32)  -- always used
//  [1024, 50176)    : Wbf  bf16[3][128][64]            -- fast path only
//  [50176, 56320)   : Agbf bf16[3][32][32] (A^T, padded)
//  [56320, 56832)   : sumb f32[128]
#define WS_W_BYTE    1024
#define WS_AG_BYTE   50176
#define WS_SB_BYTE   56320
#define WS_NEED      56832

// ---- LDS layout -----------------------------------------------------------
//  Wl  bf16[3][128][64]        @ 0       (49152 B)
//  Agl bf16[3][32][32]         @ 49152   ( 6144 B)
//  xlf f32 [64][TT][32]        @ 55296   (32768 B)  (w padded with 0)
//  xal bf16[3][TT][32][64]     @ 88064   (49152 B)  (c-fastest)
//  sumbl f32[128]              @ 137216  (  512 B)
#define LDS_W_BYTE   0
#define LDS_AG_BYTE  49152
#define LDS_X_BYTE   55296
#define LDS_XA_BYTE  88064
#define LDS_SB_BYTE  137216
#define SMEM_BYTES   137728

__device__ __forceinline__ v8bf cvt8(const float* p) {
    float4 a = *(const float4*)p;
    float4 c = *(const float4*)(p + 4);
    v8bf r;
    r[0]=(bf16)a.x; r[1]=(bf16)a.y; r[2]=(bf16)a.z; r[3]=(bf16)a.w;
    r[4]=(bf16)c.x; r[5]=(bf16)c.y; r[6]=(bf16)c.z; r[7]=(bf16)c.w;
    return r;
}

// ---- one-time constant prep: f32 -> bf16 into d_ws ------------------------
__global__ __launch_bounds__(256) void prep_consts(
    const float* __restrict__ A, const float* __restrict__ W,
    const float* __restrict__ b, char* __restrict__ ws)
{
    bf16*  Wbf  = (bf16*)(ws + WS_W_BYTE);
    bf16*  Agbf = (bf16*)(ws + WS_AG_BYTE);
    float* sumb = (float*)(ws + WS_SB_BYTE);
    int i = blockIdx.x * 256 + threadIdx.x;          // grid covers 24576
    if (i < PP*COUT*CIN) Wbf[i] = (bf16)W[i];
    if (i < PP*VP*VP) {
        int p = i >> 10, rem = i & 1023, v = rem >> 5, w = rem & 31;
        float val = (v < VV && w < VV) ? A[(p*VV + w)*VV + v] : 0.0f;
        Agbf[i] = (bf16)val;
    }
    if (i < COUT) sumb[i] = b[i] + b[COUT + i] + b[2*COUT + i];
}

template<bool FAST>
__global__ __launch_bounds__(256) void sgcn_main(
    const float* __restrict__ x,   // (N, Cin, T, V)
    const float* __restrict__ A,   // (P, V, V)          (slow path)
    const float* __restrict__ W,   // (P, Cout, Cin)     (slow path)
    const float* __restrict__ b,   // (P, Cout)          (slow path)
    const char* __restrict__ ws,   // prepped constants  (fast path)
    float* __restrict__ out)       // (N, Cout, T, V)  pre-BN result
{
    extern __shared__ char smem_raw[];
    bf16*  Wl    = (bf16*)(smem_raw + LDS_W_BYTE);
    bf16*  Agl   = (bf16*)(smem_raw + LDS_AG_BYTE);
    float* xlf   = (float*)(smem_raw + LDS_X_BYTE);
    bf16*  xal   = (bf16*)(smem_raw + LDS_XA_BYTE);
    float* sumbl = (float*)(smem_raw + LDS_SB_BYTE);

    const int tid  = threadIdx.x;
    const int lane = tid & 31;
    const int wave = tid >> 5;          // 0..7
    const int lr   = lane & 15;         // row/col within 16-tile
    const int half = lane >> 4;         // 0/1: selects K half per WMMA layout

    const int n     = blockIdx.x / (TSTEPS / TT);
    const int tbase = (blockIdx.x % (TSTEPS / TT)) * TT;

    // ---------------- Phase 0: stage operands into LDS ---------------------
    if constexpr (FAST) {
        const unsigned ldsW  = (unsigned)(uintptr_t)Wl;
        const unsigned ldsAg = (unsigned)(uintptr_t)Agl;
        const unsigned ldsX  = (unsigned)(uintptr_t)xlf;
        const char* Agg = ws + WS_AG_BYTE;

        // --- W panel (49152 contiguous bytes): one Tensor-Data-Mover DMA ---
#if defined(__has_builtin) && __has_builtin(__builtin_amdgcn_tensor_load_to_lds)
        if (tid < 32) {                       // wave 0 issues the descriptor
            unsigned long long ga = (unsigned long long)(uintptr_t)(ws + WS_W_BYTE);
            // D# group0: count=1 | lds_addr | global_addr[56:0] | type=2
            v4u g0 = { 1u, ldsW, (unsigned)ga,
                       (unsigned)(ga >> 32) | (2u << 30) };
            // D# group1: data_size=8B; tensor_dim0=tile_dim0=6144 elems; 1 row.
            v8i g1 = { (int)(3u << 16),                    // mask=0, data_size=3(8B)
                       (int)((6144u & 0xFFFFu) << 16),     // tensor_dim0[15:0]
                       (int)(1u << 16),                    // dim0 hi=0 | tensor_dim1=1
                       (int)(6144u << 16),                 // dim1 hi=0 | tile_dim0
                       1,                                  // tile_dim1=1, tile_dim2=0
                       6144, 0, 0 };                       // tensor_dim0_stride
            v4i gz = { 0, 0, 0, 0 };
#if __clang_major__ >= 23
            v8i gz8 = { 0, 0, 0, 0, 0, 0, 0, 0 };
            __builtin_amdgcn_tensor_load_to_lds(g0, g1, gz, gz, gz8, 0);
#else
            __builtin_amdgcn_tensor_load_to_lds(g0, g1, gz, gz, 0);
#endif
        }
#else
        {   // fallback: async b128 copies of the W panel
            const char* Wg = ws + WS_W_BYTE;
            #pragma unroll 4
            for (int i = tid; i < 3072; i += 256) {
                unsigned lo = ldsW + i*16, go = i*16;
                asm volatile("global_load_async_to_lds_b128 %0, %1, %2"
                             :: "v"(lo), "v"(go), "s"(Wg) : "memory");
            }
        }
#endif
        // --- Ag panel: async b128 DMA (ASYNCcnt) ---
        for (int i = tid; i < 384; i += 256) {           // 6144 B / 16
            unsigned lo = ldsAg + i*16, go = i*16;
            asm volatile("global_load_async_to_lds_b128 %0, %1, %2"
                         :: "v"(lo), "v"(go), "s"(Agg) : "memory");
        }
        // --- x tile: 25-f32 rows are only 4B-aligned -> async b32, as f32 ---
        #pragma unroll 5
        for (int i = tid; i < CIN*TT*VV; i += 256) {     // 6400 = 25*256
            int c = i / (TT*VV), rem = i - c*(TT*VV);
            int t = rem / VV,    w   = rem - t*VV;
            unsigned lo = ldsX + (((c*TT + t)*VP) + w) * 4;
            unsigned go = (((n*CIN + c)*TSTEPS + tbase + t)*VV + w) * 4;
            asm volatile("global_load_async_to_lds_b32 %0, %1, %2"
                         :: "v"(lo), "v"(go), "s"(x) : "memory");
        }
        for (int i = tid; i < CIN*TT*(VP-VV); i += 256) { // zero the w pad
            int c = i / (TT*(VP-VV)), rem = i - c*(TT*(VP-VV));
            int t = rem / (VP-VV),    w   = VV + rem - t*(VP-VV);
            xlf[(c*TT + t)*VP + w] = 0.0f;
        }
        if (tid < COUT) sumbl[tid] = ((const float*)(ws + WS_SB_BYTE))[tid];
        asm volatile("s_wait_asynccnt 0" ::: "memory");
#if defined(__has_builtin) && __has_builtin(__builtin_amdgcn_s_wait_tensorcnt)
        __builtin_amdgcn_s_wait_tensorcnt(0);
#else
        asm volatile("s_wait_tensorcnt 0" ::: "memory");
#endif
    } else {
        for (int i = tid; i < PP*COUT*CIN; i += 256)
            Wl[i] = (bf16)W[i];
        for (int i = tid; i < PP*VP*VP; i += 256) {
            int p = i >> 10, rem = i & 1023, v = rem >> 5, w = rem & 31;
            float val = (v < VV && w < VV) ? A[(p*VV + w)*VV + v] : 0.0f;
            Agl[i] = (bf16)val;
        }
        for (int i = tid; i < CIN*TT*VP; i += 256) {
            int c = i >> 7, rem = i & 127, t = rem >> 5, w = rem & 31;
            xlf[i] = (w < VV) ? x[((n*CIN + c)*TSTEPS + tbase + t)*VV + w] : 0.0f;
        }
        if (tid < COUT) sumbl[tid] = b[tid] + b[COUT + tid] + b[2*COUT + tid];
    }
    __syncthreads();

    // ---------------- Phase 1: xa_p[c, t, v] = x[c,t,:] @ A_p --------------
    // wave -> (ctile 0..3, vtile 0..1). One WMMA per (t, p). K = w (25 pad 32).
    {
        const int c0   = (wave & 3) * 16;
        const int vrow = (wave >> 2) * 16 + lr;             // B column / D column
        v16bf bg[PP];
        #pragma unroll
        for (int p = 0; p < PP; ++p)                        // B operand: K=w ascending
            bg[p] = *(const v16bf*)(Agl + ((p*VP + vrow)*VP + half*16));

        #pragma unroll
        for (int t = 0; t < TT; ++t) {
            const float* xp = xlf + (((c0 + lr)*TT + t)*VP);
            V16 ax;                                         // A operand: row c, K chunks
            ax.h[0] = cvt8(xp + half*16);                   // K = half*16 .. +7
            ax.h[1] = cvt8(xp + half*16 + 8);               // K = half*16+8 .. +15
            #pragma unroll
            for (int p = 0; p < PP; ++p) {
                v8f acc = {0,0,0,0,0,0,0,0};
                acc = __builtin_amdgcn_wmma_f32_16x16x32_bf16(
                        false, ax.v, false, bg[p], (short)0, acc, false, false);
                v8bf pk;                                    // repack D -> bf16, c-fastest
                #pragma unroll
                for (int r = 0; r < 8; ++r) pk[r] = (bf16)acc[r];
                *(v8bf*)(xal + (((p*TT + t)*VP + vrow)*CIN + c0 + half*8)) = pk;
            }
        }
    }
    __syncthreads();

    // ---------------- Phase 2: y[o, t, v] = sum_p sum_c W_p[o,c] xa_p[c,t,v]
    // wave -> o-tile. W operand chunks hoisted (3p x 2 ksteps). 6 WMMA per D tile.
    {
        const int o0 = wave * 16;
        float biasr[8];
        #pragma unroll
        for (int r = 0; r < 8; ++r) biasr[r] = sumbl[o0 + half*8 + r];

        V16 wa[PP][2];
        #pragma unroll
        for (int p = 0; p < PP; ++p)
            #pragma unroll
            for (int s = 0; s < 2; ++s) {
                const bf16* wp = Wl + ((p*COUT + o0 + lr)*CIN + s*32 + half*16);
                wa[p][s].h[0] = *(const v8bf*)(wp);
                wa[p][s].h[1] = *(const v8bf*)(wp + 8);
            }

        #pragma unroll
        for (int t = 0; t < TT; ++t) {
            #pragma unroll
            for (int vt = 0; vt < 2; ++vt) {
                v8f acc = {0,0,0,0,0,0,0,0};
                #pragma unroll
                for (int p = 0; p < PP; ++p)
                    #pragma unroll
                    for (int s = 0; s < 2; ++s) {
                        v16bf xb = *(const v16bf*)(xal +
                            (((p*TT + t)*VP + vt*16 + lr)*CIN + s*32 + half*16));
                        acc = __builtin_amdgcn_wmma_f32_16x16x32_bf16(
                                false, wa[p][s].v, false, xb, (short)0, acc, false, false);
                    }
                const int v  = vt*16 + lr;
                const int tg = tbase + t;
                if (v < VV) {
                    float* yp = out + ((n*COUT + o0 + half*8)*TSTEPS + tg)*VV + v;
                    #pragma unroll
                    for (int r = 0; r < 8; ++r)
                        yp[r * (TSTEPS*VV)] = acc[r] + biasr[r];
                }
            }
        }
    }
}

// -------- BN pass 1: per-channel mean/var -> scale/shift in d_ws[0..1024) --
__global__ __launch_bounds__(256) void bn_stats(
    const float* __restrict__ y, const float* __restrict__ gamma,
    const float* __restrict__ beta, float* __restrict__ ws)
{
    const int o = blockIdx.x;                     // 0..127
    const int TVN = TSTEPS*VV;                    // 7500
    float s = 0.0f, ss = 0.0f;
    for (int i = threadIdx.x; i < NN*TVN; i += 256) {
        int nn = i / TVN, j = i - nn*TVN;
        float v = y[(nn*COUT + o)*TVN + j];
        s += v; ss += v*v;
    }
    __shared__ float rs[256], rss[256];
    rs[threadIdx.x] = s; rss[threadIdx.x] = ss;
    __syncthreads();
    for (int st = 128; st > 0; st >>= 1) {
        if (threadIdx.x < st) {
            rs[threadIdx.x]  += rs[threadIdx.x + st];
            rss[threadIdx.x] += rss[threadIdx.x + st];
        }
        __syncthreads();
    }
    if (threadIdx.x == 0) {
        const float inv = 1.0f / (float)(NN*TVN);
        float mean = rs[0] * inv;
        float var  = rss[0] * inv - mean*mean;
        float sc   = gamma[o] * rsqrtf(var + 1e-5f);
        ws[o]        = sc;
        ws[COUT + o] = beta[o] - mean * sc;
    }
}

// -------- BN pass 2: in-place normalize + ReLU (float4, 7500 % 4 == 0) ----
__global__ __launch_bounds__(256) void bn_apply(float* __restrict__ y,
                                                const float* __restrict__ ws)
{
    const int i4 = blockIdx.x * 256 + threadIdx.x;
    const int total4 = NN*COUT*TSTEPS*VV/4;       // 7,680,000
    if (i4 >= total4) return;
    const int o = (i4 / (TSTEPS*VV/4)) % COUT;    // 1875 float4 per channel slab
    const float sc = ws[o], sh = ws[COUT + o];
    float4 v = ((float4*)y)[i4];
    v.x = fmaxf(v.x*sc + sh, 0.0f);
    v.y = fmaxf(v.y*sc + sh, 0.0f);
    v.z = fmaxf(v.z*sc + sh, 0.0f);
    v.w = fmaxf(v.w*sc + sh, 0.0f);
    ((float4*)y)[i4] = v;
}

extern "C" void kernel_launch(void* const* d_in, const int* in_sizes, int n_in,
                              void* d_out, int out_size, void* d_ws, size_t ws_size,
                              hipStream_t stream) {
    const float* x     = (const float*)d_in[0];
    const float* A     = (const float*)d_in[1];
    const float* W     = (const float*)d_in[2];
    const float* b     = (const float*)d_in[3];
    const float* gamma = (const float*)d_in[4];
    const float* beta  = (const float*)d_in[5];
    float* out = (float*)d_out;
    char*  ws  = (char*)d_ws;

    const bool fast = (ws_size >= WS_NEED);

    (void)hipFuncSetAttribute((const void*)sgcn_main<true>,
                              hipFuncAttributeMaxDynamicSharedMemorySize, SMEM_BYTES);
    (void)hipFuncSetAttribute((const void*)sgcn_main<false>,
                              hipFuncAttributeMaxDynamicSharedMemorySize, SMEM_BYTES);

    dim3 grid1(NN * (TSTEPS / TT));   // 32 * 75 = 2400 workgroups
    if (fast) {
        prep_consts<<<dim3(96), dim3(256), 0, stream>>>(A, W, b, ws);
        sgcn_main<true><<<grid1, dim3(256), SMEM_BYTES, stream>>>(
            x, A, W, b, ws, out);
    } else {
        sgcn_main<false><<<grid1, dim3(256), SMEM_BYTES, stream>>>(
            x, A, W, b, ws, out);
    }

    bn_stats<<<dim3(COUT), dim3(256), 0, stream>>>(out, gamma, beta, (float*)ws);

    const int total4 = NN*COUT*TSTEPS*VV/4;
    bn_apply<<<dim3((total4 + 255) / 256), dim3(256), 0, stream>>>(out, (float*)ws);
}
